// DLTC_33217277067616
// MI455X (gfx1250) — matrix operand order repaired and checked by reading, not compiled, not run on hardware
//
#include <hip/hip_runtime.h>
#include <hip/hip_bf16.h>

#define BB 16384
#define DD 256
#define MLM 10

typedef __attribute__((ext_vector_type(16))) _Float16 v16h;
typedef __attribute__((ext_vector_type(8)))  _Float16 v8h;
typedef __attribute__((ext_vector_type(8)))  float    v8f;

union V16HU { v16h v; v8h h[2]; };

// ---------- WMMA helpers (layouts per CDNA5 ISA 7.12.2) ----------

__device__ __forceinline__ v8f wmma_f16(v16h a, v16h b, v8f c) {
  return __builtin_amdgcn_wmma_f32_16x16x32_f16(false, a, false, b, (short)0, c,
                                                false, false);
}

// A fragment 16x32 f16, row-major source, ld elements per row.
// lane row = lane&15 ; elems 0..7 -> kBase+8*half+0..7,
// elems 8..15 -> kBase+16+8*half+0..7
__device__ __forceinline__ v16h ldA(const _Float16* base, int ld, int row,
                                    int kBase, int lane) {
  int half = lane >> 4;
  const _Float16* p = base + (size_t)row * ld + kBase + 8 * half;
  V16HU u;
  u.h[0] = *(const v8h*)(p);
  u.h[1] = *(const v8h*)(p + 16);
  return u.v;
}

// A fragment for a logical row that is concat(s0[0:256], s1[0:256]); kBase
// multiple of 32 so each 8-half chunk never straddles the 256 boundary.
__device__ __forceinline__ v16h ldA_cat(const _Float16* s0, const _Float16* s1,
                                        int kBase, int lane) {
  int half = lane >> 4;
  int k0 = kBase + 8 * half;
  int k1 = k0 + 16;
  const _Float16* p0 = (k0 < 256) ? (s0 + k0) : (s1 + (k0 - 256));
  const _Float16* p1 = (k1 < 256) ? (s0 + k1) : (s1 + (k1 - 256));
  V16HU u;
  u.h[0] = *(const v8h*)p0;
  u.h[1] = *(const v8h*)p1;
  return u.v;
}

// B fragment 32x16 f16 from row-major weight W[N,K] (computing X @ W^T):
// lane col n = nBase + (lane&15); elems 0..7 -> K = kBase+8*half+0..7,
// elems 8..15 -> K = kBase+16+8*half+0..7.
__device__ __forceinline__ v16h ldB(const _Float16* W, int ld, int nBase,
                                    int kBase, int lane) {
  int half = lane >> 4;
  int n = nBase + (lane & 15);
  const _Float16* p = W + (size_t)n * ld + kBase + 8 * half;
  V16HU u;
  u.h[0] = *(const v8h*)(p);
  u.h[1] = *(const v8h*)(p + 16);
  return u.v;
}

__device__ __forceinline__ float sigf(float x) {
  return 1.0f / (1.0f + __expf(-x));
}

// ---------- kernels ----------

__global__ void k_cvt_f16(const float* __restrict__ s, _Float16* __restrict__ d,
                          long n) {
  long i = (long)blockIdx.x * blockDim.x + threadIdx.x;
  long st = (long)gridDim.x * blockDim.x;
  for (; i < n; i += st) d[i] = (_Float16)s[i];
}

// Per-row frame weighting: norms, cosine, tiny 3->D MLP, x = [O_prev, O_t*W_t]
__global__ __launch_bounds__(256) void k_stats(const float* __restrict__ Ot,
                                               const float* __restrict__ Op,
                                               const float* __restrict__ mW,
                                               const float* __restrict__ mb,
                                               float* __restrict__ Wt_out,
                                               _Float16* __restrict__ xh) {
  int lane = threadIdx.x & 31;
  int wv = threadIdx.x >> 5;
  int b = blockIdx.x * 8 + wv;
  const float* ot = Ot + (size_t)b * DD;
  const float* op = Op + (size_t)b * DD;
  float va[8], vp[8];
  float stt = 0.f, spp = 0.f, sdd = 0.f, stp = 0.f;
#pragma unroll
  for (int i = 0; i < 8; ++i) {
    float a = ot[lane + 32 * i];
    float p = op[lane + 32 * i];
    va[i] = a; vp[i] = p;
    stt += a * a; spp += p * p;
    float dd = a - p;
    sdd += dd * dd; stp += a * p;
  }
#pragma unroll
  for (int off = 16; off; off >>= 1) {
    stt += __shfl_xor(stt, off, 32);
    spp += __shfl_xor(spp, off, 32);
    sdd += __shfl_xor(sdd, off, 32);
    stp += __shfl_xor(stp, off, 32);
  }
  float nt = sqrtf(stt), np = sqrtf(spp), rd = sqrtf(sdd);
  float ra = stp / (nt * np + 1e-6f);
#pragma unroll
  for (int i = 0; i < 8; ++i) {
    int d = lane + 32 * i;
    float w = tanhf(nt * mW[d * 3 + 0] + rd * mW[d * 3 + 1] + ra * mW[d * 3 + 2] + mb[d]);
    w = (w + 1.0f) * 0.5f;
    Wt_out[(size_t)b * DD + d] = w;
    xh[(size_t)b * 512 + d] = (_Float16)vp[i];
    xh[(size_t)b * 512 + 256 + d] = (_Float16)(va[i] * w);
  }
}

// LSTM gates GEMM, MT=2: [B,512] x Wih^T; i,g,o gates only (f*c0=0).
// A shared across 3 gates; B fragments shared across 2 M-subtiles.
__global__ __launch_bounds__(512) void k_gates(const _Float16* __restrict__ xh,
                                               const _Float16* __restrict__ Wih,
                                               const float* __restrict__ bih,
                                               const float* __restrict__ bhh,
                                               _Float16* __restrict__ otp) {
  int lane = threadIdx.x & 31;
  int wave = threadIdx.x >> 5;   // 16 waves -> 16 N-tiles (d columns)
  int mBase = blockIdx.x * 32;
  int nBase = wave * 16;
  v8f ai[2] = {{}, {}}, ag[2] = {{}, {}}, ao[2] = {{}, {}};
  for (int k = 0; k < 512; k += 32) {
    __builtin_prefetch(xh + (size_t)(mBase + (lane & 15)) * 512 + ((k + 32) & 511), 0, 3);
    v16h bi = ldB(Wih, 512, nBase + 0,   k, lane);
    v16h bg = ldB(Wih, 512, nBase + 512, k, lane);
    v16h bo = ldB(Wih, 512, nBase + 768, k, lane);
#pragma unroll
    for (int t = 0; t < 2; ++t) {
      v16h a = ldA(xh, 512, mBase + 16 * t + (lane & 15), k, lane);
      ai[t] = wmma_f16(a, bi, ai[t]);
      ag[t] = wmma_f16(a, bg, ag[t]);
      ao[t] = wmma_f16(a, bo, ao[t]);
    }
  }
  int half = lane >> 4;
  int col = nBase + (lane & 15);
  float bI = bih[col] + bhh[col];
  float bG = bih[col + 512] + bhh[col + 512];
  float bO = bih[col + 768] + bhh[col + 768];
#pragma unroll
  for (int t = 0; t < 2; ++t)
#pragma unroll
    for (int j = 0; j < 8; ++j) {
      int m = mBase + 16 * t + 8 * half + j;
      float ig = sigf(ai[t][j] + bI);
      float gg = tanhf(ag[t][j] + bG);
      float og = sigf(ao[t][j] + bO);
      otp[(size_t)m * DD + col] = (_Float16)(og * tanhf(ig * gg));
    }
}

// Q GEMM, MT=4: O' [B,256] x qW^T -> Q f32 (B fragment reused 4x)
__global__ __launch_bounds__(512) void k_q(const _Float16* __restrict__ A,
                                           const _Float16* __restrict__ W,
                                           const float* __restrict__ bias,
                                           float* __restrict__ out) {
  int lane = threadIdx.x & 31;
  int wave = threadIdx.x >> 5;
  int mBase = blockIdx.x * 64;
  int nBase = wave * 16;
  v8f acc[4] = {{}, {}, {}, {}};
  for (int k = 0; k < 256; k += 32) {
    v16h b = ldB(W, 256, nBase, k, lane);
#pragma unroll
    for (int t = 0; t < 4; ++t)
      acc[t] = wmma_f16(ldA(A, 256, mBase + 16 * t + (lane & 15), k, lane), b, acc[t]);
  }
  int half = lane >> 4;
  int col = nBase + (lane & 15);
  float bv = bias[col];
#pragma unroll
  for (int t = 0; t < 4; ++t)
#pragma unroll
    for (int j = 0; j < 8; ++j) {
      int m = mBase + 16 * t + 8 * half + j;
      out[(size_t)m * DD + col] = acc[t][j] + bv;
    }
}

// K and V GEMMs fused, MT=2: memory [B*ML,256] x {kW,vW}^T -> f16
__global__ __launch_bounds__(512) void k_kv(const _Float16* __restrict__ memh,
                                            const _Float16* __restrict__ kW,
                                            const _Float16* __restrict__ vW,
                                            const float* __restrict__ kb,
                                            const float* __restrict__ vb,
                                            _Float16* __restrict__ Kh,
                                            _Float16* __restrict__ Vh) {
  int lane = threadIdx.x & 31;
  int wave = threadIdx.x >> 5;
  int mBase = blockIdx.x * 32;
  int nBase = wave * 16;
  v8f ak[2] = {{}, {}}, av[2] = {{}, {}};
  for (int k = 0; k < 256; k += 32) {
    v16h bk_ = ldB(kW, 256, nBase, k, lane);
    v16h bv_ = ldB(vW, 256, nBase, k, lane);
#pragma unroll
    for (int t = 0; t < 2; ++t) {
      v16h a = ldA(memh, 256, mBase + 16 * t + (lane & 15), k, lane);
      ak[t] = wmma_f16(a, bk_, ak[t]);
      av[t] = wmma_f16(a, bv_, av[t]);
    }
  }
  int half = lane >> 4;
  int col = nBase + (lane & 15);
  float bk = kb[col], bv = vb[col];
#pragma unroll
  for (int t = 0; t < 2; ++t)
#pragma unroll
    for (int j = 0; j < 8; ++j) {
      size_t idx = (size_t)(mBase + 16 * t + 8 * half + j) * DD + col;
      Kh[idx] = (_Float16)(ak[t][j] + bk);
      Vh[idx] = (_Float16)(av[t][j] + bv);
    }
}

// Attention over M=10: one wave per batch row.
__global__ __launch_bounds__(256) void k_attn(const float* __restrict__ Q,
                                              const _Float16* __restrict__ Kh,
                                              const _Float16* __restrict__ Vh,
                                              float* __restrict__ out_up1,
                                              float* __restrict__ out_mem,
                                              float* __restrict__ out_up3,
                                              _Float16* __restrict__ ouph) {
  int lane = threadIdx.x & 31;
  int wv = threadIdx.x >> 5;
  int b = blockIdx.x * 8 + wv;
  const float* q = Q + (size_t)b * DD;
  float qr[8];
#pragma unroll
  for (int i = 0; i < 8; ++i) qr[i] = q[lane + 32 * i];
  float sc[MLM];
  for (int m = 0; m < MLM; ++m) {
    const _Float16* kp = Kh + ((size_t)b * MLM + m) * DD;
    float s = 0.f;
#pragma unroll
    for (int i = 0; i < 8; ++i) s += qr[i] * (float)kp[lane + 32 * i];
#pragma unroll
    for (int off = 16; off; off >>= 1) s += __shfl_xor(s, off, 32);
    sc[m] = s * 0.0625f;  // 1/sqrt(256)
  }
  float mx = sc[0];
  for (int m = 1; m < MLM; ++m) mx = fmaxf(mx, sc[m]);
  float den = 0.f;
  for (int m = 0; m < MLM; ++m) { sc[m] = __expf(sc[m] - mx); den += sc[m]; }
  float inv = 1.0f / den;
  float o[8] = {0.f, 0.f, 0.f, 0.f, 0.f, 0.f, 0.f, 0.f};
  for (int m = 0; m < MLM; ++m) {
    const _Float16* vp = Vh + ((size_t)b * MLM + m) * DD;
    float w = sc[m] * inv;
#pragma unroll
    for (int i = 0; i < 8; ++i) o[i] += w * (float)vp[lane + 32 * i];
  }
#pragma unroll
  for (int i = 0; i < 8; ++i) {
    int d = lane + 32 * i;
    float v = o[i];
    out_up1[(size_t)b * DD + d] = v;
    out_up3[(size_t)b * DD + d] = v;
    out_mem[((size_t)b * MLM + MLM - 1) * DD + d] = v;   // appended last slot
    ouph[(size_t)b * DD + d] = (_Float16)v;
  }
}

// z/r gates fused, MT=2: A = concat(memory, O_up) [B*ML,512];
// epilogue produces z (f16) and r*memory (f16).
__global__ __launch_bounds__(512) void k_zr(const _Float16* __restrict__ memh,
                                            const _Float16* __restrict__ ouph,
                                            const _Float16* __restrict__ zW,
                                            const _Float16* __restrict__ rW,
                                            const float* __restrict__ zb,
                                            const float* __restrict__ rb,
                                            const float* __restrict__ memf,
                                            _Float16* __restrict__ zh,
                                            _Float16* __restrict__ rmh) {
  int lane = threadIdx.x & 31;
  int wave = threadIdx.x >> 5;
  int mBase = blockIdx.x * 32;
  int nBase = wave * 16;
  const _Float16* s0[2];
  const _Float16* s1[2];
#pragma unroll
  for (int t = 0; t < 2; ++t) {
    int R = mBase + 16 * t + (lane & 15);
    s0[t] = memh + (size_t)R * DD;
    s1[t] = ouph + (size_t)(R / MLM) * DD;
  }
  v8f az[2] = {{}, {}}, ar[2] = {{}, {}};
  for (int k = 0; k < 512; k += 32) {
    v16h bz_ = ldB(zW, 512, nBase, k, lane);
    v16h br_ = ldB(rW, 512, nBase, k, lane);
#pragma unroll
    for (int t = 0; t < 2; ++t) {
      v16h a = ldA_cat(s0[t], s1[t], k, lane);
      az[t] = wmma_f16(a, bz_, az[t]);
      ar[t] = wmma_f16(a, br_, ar[t]);
    }
  }
  int half = lane >> 4;
  int col = nBase + (lane & 15);
  float bz = zb[col], br = rb[col];
#pragma unroll
  for (int t = 0; t < 2; ++t)
#pragma unroll
    for (int j = 0; j < 8; ++j) {
      size_t row = (size_t)(mBase + 16 * t + 8 * half + j);
      float z = sigf(az[t][j] + bz);
      float r = sigf(ar[t][j] + br);
      zh[row * DD + col] = (_Float16)z;
      rmh[row * DD + col] = (_Float16)(r * memf[row * DD + col]);
    }
}

// h gate + GRU blend + shifted store, MT=4: A = concat(r*memory, O_up).
__global__ __launch_bounds__(512) void k_h(const _Float16* __restrict__ rmh,
                                           const _Float16* __restrict__ ouph,
                                           const _Float16* __restrict__ hW,
                                           const float* __restrict__ hb,
                                           const _Float16* __restrict__ zh,
                                           const float* __restrict__ memf,
                                           float* __restrict__ out_mem) {
  int lane = threadIdx.x & 31;
  int wave = threadIdx.x >> 5;
  int mBase = blockIdx.x * 64;
  int nBase = wave * 16;
  const _Float16* s0[4];
  const _Float16* s1[4];
#pragma unroll
  for (int t = 0; t < 4; ++t) {
    int R = mBase + 16 * t + (lane & 15);
    s0[t] = rmh + (size_t)R * DD;
    s1[t] = ouph + (size_t)(R / MLM) * DD;
  }
  v8f acc[4] = {{}, {}, {}, {}};
  for (int k = 0; k < 512; k += 32) {
    v16h b = ldB(hW, 512, nBase, k, lane);
#pragma unroll
    for (int t = 0; t < 4; ++t)
      acc[t] = wmma_f16(ldA_cat(s0[t], s1[t], k, lane), b, acc[t]);
  }
  int half = lane >> 4;
  int col = nBase + (lane & 15);
  float bh = hb[col];
#pragma unroll
  for (int t = 0; t < 4; ++t)
#pragma unroll
    for (int j = 0; j < 8; ++j) {
      int row = mBase + 16 * t + 8 * half + j;
      float mt = tanhf(acc[t][j] + bh);
      float z = (float)zh[(size_t)row * DD + col];
      float mem = memf[(size_t)row * DD + col];
      float val = z * mem + (1.0f - z) * mt;
      int mm = row % MLM;
      if (mm > 0)  // memory_new[:, 0:ML-1] = updated rows 1..ML-1 (shift-left)
        out_mem[(size_t)(row - 1) * DD + col] = val;
    }
}

// ---------- host launcher ----------

extern "C" void kernel_launch(void* const* d_in, const int* in_sizes, int n_in,
                              void* d_out, int out_size, void* d_ws, size_t ws_size,
                              hipStream_t stream) {
  const float* O_t   = (const float*)d_in[0];
  const float* O_prev= (const float*)d_in[1];
  const float* mem   = (const float*)d_in[2];
  const float* mW    = (const float*)d_in[3];
  const float* mb    = (const float*)d_in[4];
  const float* Wih   = (const float*)d_in[5];
  // d_in[6] lstm_Whh unused (h0 = 0)
  const float* bih   = (const float*)d_in[7];
  const float* bhh   = (const float*)d_in[8];
  const float* qW    = (const float*)d_in[9];
  const float* qb    = (const float*)d_in[10];
  const float* kW    = (const float*)d_in[11];
  const float* kb    = (const float*)d_in[12];
  const float* vW    = (const float*)d_in[13];
  const float* vb    = (const float*)d_in[14];
  const float* zW    = (const float*)d_in[15];
  const float* zb    = (const float*)d_in[16];
  const float* rW    = (const float*)d_in[17];
  const float* rb    = (const float*)d_in[18];
  const float* hW    = (const float*)d_in[19];
  const float* hb    = (const float*)d_in[20];
  (void)in_sizes; (void)n_in; (void)out_size; (void)ws_size;

  float* out = (float*)d_out;
  float* out_up1 = out;
  float* out_mem = out_up1 + (size_t)BB * DD;
  float* out_up3 = out_mem + (size_t)BB * MLM * DD;
  float* out_wt  = out_up3 + (size_t)BB * DD;

  char* w = (char*)d_ws;
  auto alloc = [&](size_t bytes) {
    char* p = w;
    w += (bytes + 255) & ~(size_t)255;
    return p;
  };
  _Float16* Wih_h = (_Float16*)alloc((size_t)1024 * 512 * 2);
  _Float16* qW_h  = (_Float16*)alloc((size_t)256 * 256 * 2);
  _Float16* kW_h  = (_Float16*)alloc((size_t)256 * 256 * 2);
  _Float16* vW_h  = (_Float16*)alloc((size_t)256 * 256 * 2);
  _Float16* zW_h  = (_Float16*)alloc((size_t)256 * 512 * 2);
  _Float16* rW_h  = (_Float16*)alloc((size_t)256 * 512 * 2);
  _Float16* hW_h  = (_Float16*)alloc((size_t)256 * 512 * 2);
  _Float16* x_h   = (_Float16*)alloc((size_t)BB * 512 * 2);
  _Float16* mem_h = (_Float16*)alloc((size_t)BB * MLM * DD * 2);
  _Float16* otp_h = (_Float16*)alloc((size_t)BB * DD * 2);
  float*    Qf    = (float*)   alloc((size_t)BB * DD * 4);
  _Float16* K_h   = (_Float16*)alloc((size_t)BB * MLM * DD * 2);
  _Float16* V_h   = (_Float16*)alloc((size_t)BB * MLM * DD * 2);
  _Float16* oup_h = (_Float16*)alloc((size_t)BB * DD * 2);
  _Float16* z_h   = (_Float16*)alloc((size_t)BB * MLM * DD * 2);
  _Float16* rm_h  = (_Float16*)alloc((size_t)BB * MLM * DD * 2);

  // 1) f16 conversions (weights + memory)
  k_cvt_f16<<<512, 256, 0, stream>>>(Wih, Wih_h, (long)1024 * 512);
  k_cvt_f16<<<128, 256, 0, stream>>>(qW, qW_h, (long)256 * 256);
  k_cvt_f16<<<128, 256, 0, stream>>>(kW, kW_h, (long)256 * 256);
  k_cvt_f16<<<128, 256, 0, stream>>>(vW, vW_h, (long)256 * 256);
  k_cvt_f16<<<256, 256, 0, stream>>>(zW, zW_h, (long)256 * 512);
  k_cvt_f16<<<256, 256, 0, stream>>>(rW, rW_h, (long)256 * 512);
  k_cvt_f16<<<256, 256, 0, stream>>>(hW, hW_h, (long)256 * 512);
  k_cvt_f16<<<4096, 256, 0, stream>>>(mem, mem_h, (long)BB * MLM * DD);

  // 2) frame weighting -> W_t (output) + x (f16)
  k_stats<<<BB / 8, 256, 0, stream>>>(O_t, O_prev, mW, mb, out_wt, x_h);

  // 3) LSTM gates GEMM (WMMA, MT=2) -> O' (f16)
  k_gates<<<BB / 32, 512, 0, stream>>>(x_h, Wih_h, bih, bhh, otp_h);

  // 4) Q, K, V GEMMs (WMMA)
  k_q<<<BB / 64, 512, 0, stream>>>(otp_h, qW_h, qb, Qf);
  k_kv<<<(BB * MLM) / 32, 512, 0, stream>>>(mem_h, kW_h, vW_h, kb, vb, K_h, V_h);

  // 5) attention -> O_up (outputs 1 & 3, memory slot ML-1, f16 copy)
  k_attn<<<BB / 8, 256, 0, stream>>>(Qf, K_h, V_h, out_up1, out_mem, out_up3, oup_h);

  // 6) z/r gates GEMM (WMMA, MT=2) -> z (f16), r*memory (f16)
  k_zr<<<(BB * MLM) / 32, 512, 0, stream>>>(mem_h, oup_h, zW_h, rW_h, zb, rb,
                                            mem, z_h, rm_h);

  // 7) h gate GEMM (WMMA, MT=4) + GRU blend + shifted memory_new store
  k_h<<<(BB * MLM) / 64, 512, 0, stream>>>(rm_h, oup_h, hW_h, hb, z_h, mem, out_mem);
}
